// RelativePositionAttention_38998303047754
// MI455X (gfx1250) — matrix operand order repaired and checked by reading, not compiled, not run on hardware
//
#include <hip/hip_runtime.h>
#include <hip/hip_bf16.h>

// ---------------------------------------------------------------------------
// Relative-position multi-head attention for MI455X (gfx1250), wave32 WMMA.
// All matmuls use v_wmma_f32_16x16x32_bf16 (fp32 accumulate). The attn tensor
// (537 MB) dominates HBM traffic, so compute is done in bf16 and the pipeline
// is arranged to touch attn the minimum number of times.
// ---------------------------------------------------------------------------

typedef __bf16 bf16;
typedef __attribute__((ext_vector_type(16))) __bf16 v16bf;
typedef __attribute__((ext_vector_type(8)))  __bf16 v8bf;
typedef __attribute__((ext_vector_type(8)))  float  v8f;
typedef __attribute__((ext_vector_type(4)))  float  v4f;

#define D_MODEL   1024
#define N_HEADS   16
#define HEAD_DIM  64
#define SEQ       2048
#define BATCH     2
#define MAX_REL   32
#define REL_ROWS  65          // 2*MAX_REL+1
#define REL_PAD   128         // padded N for the rel GEMM (multiple of 64)
#define SCALE     0.125f      // 1/sqrt(64)

union Frag16 { v16bf v; v8bf h[2]; };

__device__ __forceinline__ v8f wmma_bf16(v16bf a, v16bf b, v8f c) {
    // 8 args: (neg_a, A, neg_b, B, c_mod, C, reuse_a, reuse_b)
    return __builtin_amdgcn_wmma_f32_16x16x32_bf16(false, a, false, b,
                                                   (short)0, c, false, false);
}

// A fragment (16x32, bf16 source). Per ISA 7.12.2: lane l holds row (l&15);
// elements 0..7 = K (l>>4)*8..+7, elements 8..15 = K 16+(l>>4)*8..+7.
__device__ __forceinline__ v16bf load_afrag_bf16(const bf16* a, int lda,
                                                 int m, int k,
                                                 int laneLo, int laneHi) {
    const bf16* p = a + (size_t)(m + laneLo) * lda + k + laneHi * 8;
    Frag16 f;
    f.h[0] = *(const v8bf*)(p);
    f.h[1] = *(const v8bf*)(p + 16);
    return f.v;
}

// A fragment from an fp32 source: load 2x(2x float4), convert to bf16 in regs.
__device__ __forceinline__ v16bf load_afrag_f32(const float* a, int lda,
                                                int m, int k,
                                                int laneLo, int laneHi) {
    const float* p = a + (size_t)(m + laneLo) * lda + k + laneHi * 8;
    v4f x0 = *(const v4f*)(p);
    v4f x1 = *(const v4f*)(p + 4);
    v4f x2 = *(const v4f*)(p + 16);
    v4f x3 = *(const v4f*)(p + 20);
    v16bf f;
#pragma unroll
    for (int i = 0; i < 4; ++i) {
        f[i]      = (bf16)x0[i];
        f[4 + i]  = (bf16)x1[i];
        f[8 + i]  = (bf16)x2[i];
        f[12 + i] = (bf16)x3[i];
    }
    return f;
}

// B fragment (32x16). Lane l holds column (l&15); elements 0..7 = K (l>>4)*8..+7,
// elements 8..15 = K 16+(l>>4)*8..+7 (B is supplied pre-transposed, Bt[n][k]).
__device__ __forceinline__ v16bf load_bfrag(const bf16* bt, int ldb,
                                            int n, int k,
                                            int laneLo, int laneHi) {
    const bf16* p = bt + (size_t)(n + laneLo) * ldb + k + laneHi * 8;
    Frag16 f;
    f.h[0] = *(const v8bf*)(p);
    f.h[1] = *(const v8bf*)(p + 16);
    return f.v;
}

// ---------------------------------------------------------------------------
// Generic batched GEMM: C[z] = A[z] @ Bt[z]^T (+bias). Block = 64x64 tile,
// 4 waves, each wave a 32x32 tile (2x2 wmma accumulators). M,N % 64 == 0,
// K % 32 == 0 for all uses, so no bounds checks.
// ---------------------------------------------------------------------------
template<bool A_F32, bool OUT_BF16, bool BIAS>
__global__ __launch_bounds__(128) void k_gemm(
        const void* __restrict__ Av, const bf16* __restrict__ Btv,
        const float* __restrict__ bias, void* __restrict__ Cv,
        int K, int lda, int ldb, int ldc,
        long long aB, long long bB, long long cB) {
    const long long z = blockIdx.z;
    const bf16* Bt = Btv + z * bB;
    const float* Af = (const float*)Av + (A_F32 ? z * aB : 0);
    const bf16*  Ab = (const bf16*)Av + (A_F32 ? 0 : z * aB);

    const int lane   = threadIdx.x & 31;
    const int laneLo = lane & 15, laneHi = lane >> 4;
    const int wv     = threadIdx.x >> 5;
    const int m0     = blockIdx.y * 64 + (wv >> 1) * 32;
    const int n0     = blockIdx.x * 64 + (wv & 1) * 32;

    v8f acc[2][2] = {};
    for (int k0 = 0; k0 < K; k0 += 32) {
        if (k0 + 32 < K) {   // global_prefetch_b8 for next k-slab
            __builtin_prefetch(Bt + (size_t)(n0 + laneLo) * ldb + k0 + 32, 0, 3);
            if (A_F32)
                __builtin_prefetch(Af + (size_t)(m0 + laneLo) * lda + k0 + 32, 0, 3);
            else
                __builtin_prefetch(Ab + (size_t)(m0 + laneLo) * lda + k0 + 32, 0, 3);
        }
        v16bf a0, a1;
        if constexpr (A_F32) {
            a0 = load_afrag_f32(Af, lda, m0,      k0, laneLo, laneHi);
            a1 = load_afrag_f32(Af, lda, m0 + 16, k0, laneLo, laneHi);
        } else {
            a0 = load_afrag_bf16(Ab, lda, m0,      k0, laneLo, laneHi);
            a1 = load_afrag_bf16(Ab, lda, m0 + 16, k0, laneLo, laneHi);
        }
        v16bf b0 = load_bfrag(Bt, ldb, n0,      k0, laneLo, laneHi);
        v16bf b1 = load_bfrag(Bt, ldb, n0 + 16, k0, laneLo, laneHi);
        acc[0][0] = wmma_bf16(a0, b0, acc[0][0]);
        acc[0][1] = wmma_bf16(a0, b1, acc[0][1]);
        acc[1][0] = wmma_bf16(a1, b0, acc[1][0]);
        acc[1][1] = wmma_bf16(a1, b1, acc[1][1]);
    }
#pragma unroll
    for (int mt = 0; mt < 2; ++mt)
#pragma unroll
        for (int nt = 0; nt < 2; ++nt) {
            const int col = n0 + nt * 16 + laneLo;
            const float bv = BIAS ? bias[col] : 0.f;
#pragma unroll
            for (int i = 0; i < 8; ++i) {
                const int row = m0 + mt * 16 + laneHi * 8 + i;
                const float vv = acc[mt][nt][i] + bv;
                const size_t idx = (size_t)(z * cB) + (size_t)row * ldc + col;
                if constexpr (OUT_BF16) ((bf16*)Cv)[idx] = (bf16)vv;
                else                    ((float*)Cv)[idx] = vv;
            }
        }
}

// ---------------------------------------------------------------------------
// Scores: per (b,h): q @ k^T (K=64) + gather(qrel, clamp(j-i)) then * SCALE.
// Writes raw (pre-softmax) fp32 scores straight into the attn output region.
// ---------------------------------------------------------------------------
__global__ __launch_bounds__(128) void k_scores(
        const bf16* __restrict__ Qp, const bf16* __restrict__ Kp,
        const bf16* __restrict__ qrel, float* __restrict__ attn) {
    const int z = blockIdx.z, b = z >> 4, h = z & (N_HEADS - 1);
    const bf16* A = Qp + (size_t)b * SEQ * D_MODEL + h * HEAD_DIM;
    const bf16* B = Kp + (size_t)b * SEQ * D_MODEL + h * HEAD_DIM;
    float* out = attn + (size_t)z * SEQ * SEQ;

    const int lane   = threadIdx.x & 31;
    const int laneLo = lane & 15, laneHi = lane >> 4;
    const int wv     = threadIdx.x >> 5;
    const int m0     = blockIdx.y * 64 + (wv >> 1) * 32;
    const int n0     = blockIdx.x * 64 + (wv & 1) * 32;

    v8f acc[2][2] = {};
#pragma unroll
    for (int k0 = 0; k0 < HEAD_DIM; k0 += 32) {
        v16bf a0 = load_afrag_bf16(A, D_MODEL, m0,      k0, laneLo, laneHi);
        v16bf a1 = load_afrag_bf16(A, D_MODEL, m0 + 16, k0, laneLo, laneHi);
        v16bf b0 = load_bfrag(B, D_MODEL, n0,      k0, laneLo, laneHi);
        v16bf b1 = load_bfrag(B, D_MODEL, n0 + 16, k0, laneLo, laneHi);
        acc[0][0] = wmma_bf16(a0, b0, acc[0][0]);
        acc[0][1] = wmma_bf16(a0, b1, acc[0][1]);
        acc[1][0] = wmma_bf16(a1, b0, acc[1][0]);
        acc[1][1] = wmma_bf16(a1, b1, acc[1][1]);
    }
#pragma unroll
    for (int mt = 0; mt < 2; ++mt)
#pragma unroll
        for (int nt = 0; nt < 2; ++nt) {
            const int col = n0 + nt * 16 + laneLo;
#pragma unroll
            for (int i = 0; i < 8; ++i) {
                const int row = m0 + mt * 16 + laneHi * 8 + i;
                int d = col - row;
                d = d < -MAX_REL ? -MAX_REL : (d > MAX_REL ? MAX_REL : d);
                d += MAX_REL;
                const float rv = (float)qrel[(((size_t)b * SEQ + row) * N_HEADS + h)
                                             * REL_PAD + d];
                out[(size_t)row * SEQ + col] = (acc[mt][nt][i] + rv) * SCALE;
            }
        }
}

// Row softmax in place: one 256-thread block per row (B*H*L rows of 2048).
__global__ __launch_bounds__(256) void k_softmax(float* __restrict__ attn) {
    float* p = attn + (size_t)blockIdx.x * SEQ;
    const int t = threadIdx.x;
    float x[SEQ / 256];
    float m = -3.0e38f;
#pragma unroll
    for (int j = 0; j < SEQ / 256; ++j) {
        x[j] = p[t + j * 256];
        m = fmaxf(m, x[j]);
    }
#pragma unroll
    for (int off = 16; off; off >>= 1) m = fmaxf(m, __shfl_xor(m, off, 32));
    __shared__ float redM[8], redS[8];
    if ((t & 31) == 0) redM[t >> 5] = m;
    __syncthreads();
    float bm = redM[0];
#pragma unroll
    for (int i = 1; i < 8; ++i) bm = fmaxf(bm, redM[i]);
    float s = 0.f;
#pragma unroll
    for (int j = 0; j < SEQ / 256; ++j) {
        x[j] = __expf(x[j] - bm);
        s += x[j];
    }
#pragma unroll
    for (int off = 16; off; off >>= 1) s += __shfl_xor(s, off, 32);
    if ((t & 31) == 0) redS[t >> 5] = s;
    __syncthreads();
    float bs = 0.f;
#pragma unroll
    for (int i = 0; i < 8; ++i) bs += redS[i];
    const float inv = 1.f / bs;
#pragma unroll
    for (int j = 0; j < SEQ / 256; ++j) p[t + j * 256] = x[j] * inv;
}

// attn(fp32, converted in-regs) @ v  via vT[b,h,ch,key]; out bf16 [B,L,D].
__global__ __launch_bounds__(128) void k_av(
        const float* __restrict__ attn, const bf16* __restrict__ vT,
        bf16* __restrict__ AO) {
    const int z = blockIdx.z, b = z >> 4, h = z & (N_HEADS - 1);
    const float* A = attn + (size_t)z * SEQ * SEQ;
    const bf16*  B = vT   + (size_t)z * HEAD_DIM * SEQ;

    const int lane   = threadIdx.x & 31;
    const int laneLo = lane & 15, laneHi = lane >> 4;
    const int wv     = threadIdx.x >> 5;
    const int m0     = blockIdx.y * 64 + (wv >> 1) * 32;
    const int n0     = (wv & 1) * 32;            // N = 64

    v8f acc[2][2] = {};
    for (int k0 = 0; k0 < SEQ; k0 += 32) {
        if (k0 + 32 < SEQ)
            __builtin_prefetch(A + (size_t)(m0 + laneLo) * SEQ + k0 + 32, 0, 3);
        v16bf a0 = load_afrag_f32(A, SEQ, m0,      k0, laneLo, laneHi);
        v16bf a1 = load_afrag_f32(A, SEQ, m0 + 16, k0, laneLo, laneHi);
        v16bf b0 = load_bfrag(B, SEQ, n0,      k0, laneLo, laneHi);
        v16bf b1 = load_bfrag(B, SEQ, n0 + 16, k0, laneLo, laneHi);
        acc[0][0] = wmma_bf16(a0, b0, acc[0][0]);
        acc[0][1] = wmma_bf16(a0, b1, acc[0][1]);
        acc[1][0] = wmma_bf16(a1, b0, acc[1][0]);
        acc[1][1] = wmma_bf16(a1, b1, acc[1][1]);
    }
#pragma unroll
    for (int mt = 0; mt < 2; ++mt)
#pragma unroll
        for (int nt = 0; nt < 2; ++nt) {
            const int col = n0 + nt * 16 + laneLo;
#pragma unroll
            for (int i = 0; i < 8; ++i) {
                const int row = m0 + mt * 16 + laneHi * 8 + i;
                AO[((size_t)b * SEQ + row) * D_MODEL + h * HEAD_DIM + col]
                    = (bf16)acc[mt][nt][i];
            }
        }
}

// Weight transpose + fp32->bf16: Wt[n][k] = W[k][n].  LDS-tiled 32x32.
__global__ void k_wt(const float* __restrict__ W, bf16* __restrict__ Wt) {
    __shared__ float t[32][33];
    const int c0 = blockIdx.x * 32, r0 = blockIdx.y * 32, tx = threadIdx.x;
    for (int j = threadIdx.y; j < 32; j += 8)
        t[j][tx] = W[(size_t)(r0 + j) * D_MODEL + c0 + tx];
    __syncthreads();
    for (int j = threadIdx.y; j < 32; j += 8)
        Wt[(size_t)(c0 + j) * D_MODEL + r0 + tx] = (bf16)t[tx][j];
}

// v transpose per head: Vp[B,L,H,dk] bf16 -> vT[B,H,dk,L] bf16.
__global__ void k_vt(const bf16* __restrict__ Vp, bf16* __restrict__ vT) {
    __shared__ bf16 t[32][33];
    const int z = blockIdx.z, b = z >> 4, h = z & (N_HEADS - 1);
    const int l0 = blockIdx.x * 32, c0 = blockIdx.y * 32, tx = threadIdx.x;
    for (int j = threadIdx.y; j < 32; j += 8)
        t[j][tx] = Vp[((size_t)b * SEQ + l0 + j) * D_MODEL + h * HEAD_DIM + c0 + tx];
    __syncthreads();
    for (int j = threadIdx.y; j < 32; j += 8)
        vT[((size_t)z * HEAD_DIM + c0 + j) * SEQ + l0 + tx] = t[tx][j];
}

// rel table [65,64] fp32 -> bf16, zero-padded to [128,64].
__global__ void k_rel(const float* __restrict__ R, bf16* __restrict__ Rp) {
    const int i = blockIdx.x * 256 + threadIdx.x;
    if (i < REL_PAD * HEAD_DIM) {
        const int r = i / HEAD_DIM;
        Rp[i] = (r < REL_ROWS) ? (bf16)R[i] : (bf16)0.f;
    }
}

// ---------------------------------------------------------------------------
extern "C" void kernel_launch(void* const* d_in, const int* in_sizes, int n_in,
                              void* d_out, int out_size, void* d_ws, size_t ws_size,
                              hipStream_t stream) {
    (void)in_sizes; (void)n_in; (void)out_size; (void)ws_size;
    const float* Q  = (const float*)d_in[0];
    const float* K  = (const float*)d_in[1];
    const float* V  = (const float*)d_in[2];
    const float* Wq = (const float*)d_in[3];
    const float* bq = (const float*)d_in[4];
    const float* Wk = (const float*)d_in[5];
    const float* bk = (const float*)d_in[6];
    const float* Wv = (const float*)d_in[7];
    const float* bv = (const float*)d_in[8];
    const float* Wo = (const float*)d_in[9];
    const float* bo = (const float*)d_in[10];
    const float* Rt = (const float*)d_in[11];

    float* out  = (float*)d_out;
    float* attn = out + (size_t)BATCH * SEQ * D_MODEL;

    // workspace (bf16 elements, ~67 MB total)
    bf16* w = (bf16*)d_ws;
    size_t o = 0;
    bf16* WqT = w + o; o += (size_t)D_MODEL * D_MODEL;
    bf16* WkT = w + o; o += (size_t)D_MODEL * D_MODEL;
    bf16* WvT = w + o; o += (size_t)D_MODEL * D_MODEL;
    bf16* WoT = w + o; o += (size_t)D_MODEL * D_MODEL;
    bf16* Rp  = w + o; o += (size_t)REL_PAD * HEAD_DIM;
    bf16* Qp  = w + o; o += (size_t)BATCH * SEQ * D_MODEL;
    bf16* Kp  = w + o; o += (size_t)BATCH * SEQ * D_MODEL;
    bf16* Vp  = w + o; o += (size_t)BATCH * SEQ * D_MODEL;
    bf16* Vt  = w + o; o += (size_t)BATCH * N_HEADS * HEAD_DIM * SEQ;
    bf16* Qr  = w + o; o += (size_t)BATCH * SEQ * N_HEADS * REL_PAD;
    bf16* AO  = w + o; o += (size_t)BATCH * SEQ * D_MODEL;

    const dim3 tb(32, 8);
    const int ML = BATCH * SEQ;                  // 4096 rows

    // 1) weight transposes + rel pad
    k_wt<<<dim3(32, 32), tb, 0, stream>>>(Wq, WqT);
    k_wt<<<dim3(32, 32), tb, 0, stream>>>(Wk, WkT);
    k_wt<<<dim3(32, 32), tb, 0, stream>>>(Wv, WvT);
    k_wt<<<dim3(32, 32), tb, 0, stream>>>(Wo, WoT);
    k_rel<<<(REL_PAD * HEAD_DIM + 255) / 256, 256, 0, stream>>>(Rt, Rp);

    // 2) projections: [4096,1024] = A_f32 @ Wt_bf16 + b, out bf16
    const dim3 gp(D_MODEL / 64, ML / 64, 1);
    k_gemm<true,  true,  true><<<gp, 128, 0, stream>>>(Q, WqT, bq, Qp,
        D_MODEL, D_MODEL, D_MODEL, D_MODEL, 0, 0, 0);
    k_gemm<true,  true,  true><<<gp, 128, 0, stream>>>(K, WkT, bk, Kp,
        D_MODEL, D_MODEL, D_MODEL, D_MODEL, 0, 0, 0);
    k_gemm<true,  true,  true><<<gp, 128, 0, stream>>>(V, WvT, bv, Vp,
        D_MODEL, D_MODEL, D_MODEL, D_MODEL, 0, 0, 0);

    // 3) v transpose to [B,H,dk,L]
    k_vt<<<dim3(SEQ / 32, HEAD_DIM / 32, BATCH * N_HEADS), tb, 0, stream>>>(Vp, Vt);

    // 4) qrel[b,l,h,:] = q_head @ Rp^T : batched over heads
    k_gemm<false, true, false><<<dim3(REL_PAD / 64, ML / 64, N_HEADS), 128, 0, stream>>>(
        Qp, Rp, nullptr, Qr,
        HEAD_DIM, D_MODEL, HEAD_DIM, N_HEADS * REL_PAD,
        /*aB=*/HEAD_DIM, /*bB=*/0, /*cB=*/REL_PAD);

    // 5) raw scores -> attn region (fp32)
    k_scores<<<dim3(SEQ / 64, SEQ / 64, BATCH * N_HEADS), 128, 0, stream>>>(
        Qp, Kp, Qr, attn);

    // 6) softmax in place
    k_softmax<<<BATCH * N_HEADS * SEQ, 256, 0, stream>>>(attn);

    // 7) attn @ v -> AO bf16 [B,L,D]
    k_av<<<dim3(1, SEQ / 64, BATCH * N_HEADS), 128, 0, stream>>>(attn, Vt, AO);

    // 8) output projection -> d_out fp32
    k_gemm<false, false, true><<<gp, 128, 0, stream>>>(AO, WoT, bo, out,
        D_MODEL, D_MODEL, D_MODEL, D_MODEL, 0, 0, 0);
}
